// DeformableTransformerDecoderLayer_17085379903979
// MI455X (gfx1250) — compile-verified, matrix operand order checked by reading, not run on hardware
//
#include <hip/hip_runtime.h>
#include <hip/hip_bf16.h>

// ---------------------------------------------------------------------------
// DeformableTransformerDecoderLayer for MI455X (gfx1250, wave32, WMMA).
// f16 operands everywhere (pre-converted), f32 WMMA accumulation.
// Async global->LDS copy (CDNA5) with sync fallback, double-buffered.
// ---------------------------------------------------------------------------

#define D_MODEL 256
#define D_FF    2048
#define H_HEADS 8
#define N_LEVELS 5
#define N_POINTS 4
#define LQ 900
#define BB 8
#define D_HEAD 32
#define LV 21824            // 128^2 + 64^2 + 32^2 + 16^2 + 8^2
#define NTOK (LQ * BB)      // 7200 tokens, row id = q*B + b  -> (Lq,B,*) layout

typedef __attribute__((ext_vector_type(16))) _Float16 v16h;
typedef __attribute__((ext_vector_type(8)))  _Float16 v8h;
typedef __attribute__((ext_vector_type(8)))  float    v8f;
typedef __attribute__((ext_vector_type(4)))  int      v4i;

union HFrag { v16h v; v8h h2[2]; _Float16 e[16]; };

__device__ __forceinline__ v8f wmma_f16(const HFrag& a, const HFrag& b, v8f c) {
  return __builtin_amdgcn_wmma_f32_16x16x32_f16(
      false, a.v, false, b.v, (short)0, c, false, false);
}

// ---- CDNA5 async global->LDS copy (guarded; sync fallback keeps compile safe)
#if defined(__has_builtin)
#if __has_builtin(__builtin_amdgcn_global_load_async_to_lds_b128) && \
    __has_builtin(__builtin_amdgcn_s_wait_asynccnt)
#define USE_ASYNC_COPY 1
#endif
#endif
#ifndef USE_ASYNC_COPY
#define USE_ASYNC_COPY 0
#endif

// param 1: generic v4i* (global/flat source)
// param 2: AS3 v4i*   (LDS destination; built via inttoptr — low 32 bits of a
//                      generic LDS pointer are the LDS offset on AMDGPU)
#define SRCV4(p) ((v4i*)(void*)(p))
#define LDSV4(p) ((__attribute__((address_space(3))) v4i*)(unsigned int)(unsigned long long)(p))

// ---------------------------------------------------------------------------
// GEMM: C[M,N] = act( (A[M,K] @ Wt[N,K]^T + bias) * scale )
// A: f16 row-major [M][K].  Wt: f16 pre-transposed [N][K].
// 128 threads / 4 waves, tile 64x64, K-stage 64 (2 WMMA substeps), ping-pong LDS.
// ---------------------------------------------------------------------------
template<bool OUT_F16, bool RELU>
__global__ __launch_bounds__(128)
void gemm_wmma(const _Float16* __restrict__ A, const _Float16* __restrict__ Wt,
               const float* __restrict__ bias, void* __restrict__ Cv,
               int M, int K, int N, float scale)
{
  __shared__ _Float16 a_s[2][64][72];   // stride 72 halves = 144B (16B aligned)
  __shared__ _Float16 b_s[2][64][72];

  const int tid  = threadIdx.x;
  const int lane = tid & 31;
  const int wave = tid >> 5;
  const int wm = (wave >> 1) * 32;
  const int wn = (wave & 1) * 32;
  const int m0 = blockIdx.y * 64;
  const int n0 = blockIdx.x * 64;
  const int l15 = lane & 15;
  const int kb  = (lane >> 4) * 8;      // K-half select per ISA 16-bit layout

  auto stage = [&](int buf, int k0) {
#pragma unroll
    for (int i = 0; i < 4; ++i) {
      int c  = i * 128 + tid;           // 512 16B-chunks per 64x64-half tile
      int r  = c >> 3;
      int co = (c & 7) * 8;
      int ga = m0 + r; if (ga >= M) ga = M - 1;
      int gb = n0 + r; if (gb >= N) gb = N - 1;
      const _Float16* srcA = A  + (size_t)ga * K + k0 + co;
      const _Float16* srcB = Wt + (size_t)gb * K + k0 + co;
#if USE_ASYNC_COPY
      __builtin_amdgcn_global_load_async_to_lds_b128(SRCV4(srcA), LDSV4(&a_s[buf][r][co]), 0, 0);
      __builtin_amdgcn_global_load_async_to_lds_b128(SRCV4(srcB), LDSV4(&b_s[buf][r][co]), 0, 0);
#else
      *(v8h*)&a_s[buf][r][co] = *(const v8h*)srcA;
      *(v8h*)&b_s[buf][r][co] = *(const v8h*)srcB;
#endif
    }
  };

  v8f acc[2][2] = {};
  stage(0, 0);
  int buf = 0;
  for (int k0 = 0; k0 < K; k0 += 64, buf ^= 1) {
#if USE_ASYNC_COPY
    __builtin_amdgcn_s_wait_asynccnt(0);
#endif
    __syncthreads();
    if (k0 + 64 < K) stage(buf ^ 1, k0 + 64);
#pragma unroll
    for (int ks = 0; ks < 2; ++ks) {
      const int kk = ks * 32 + kb;
      HFrag af[2], bf[2];
#pragma unroll
      for (int t = 0; t < 2; ++t) {
        const int mr = wm + t * 16 + l15;
        const int nr = wn + t * 16 + l15;
        af[t].h2[0] = *(const v8h*)&a_s[buf][mr][kk];
        af[t].h2[1] = *(const v8h*)&a_s[buf][mr][16 + kk];
        bf[t].h2[0] = *(const v8h*)&b_s[buf][nr][kk];
        bf[t].h2[1] = *(const v8h*)&b_s[buf][nr][16 + kk];
      }
#pragma unroll
      for (int mt = 0; mt < 2; ++mt)
#pragma unroll
        for (int nt = 0; nt < 2; ++nt)
          acc[mt][nt] = wmma_f16(af[mt], bf[nt], acc[mt][nt]);
    }
  }

  // epilogue: C layout -> VGPR i = row i (lanes 0-15) / row 8+i (lanes 16-31)
  const int rbase = (lane >> 4) * 8;
#pragma unroll
  for (int mt = 0; mt < 2; ++mt)
#pragma unroll
    for (int nt = 0; nt < 2; ++nt) {
      int col = n0 + wn + nt * 16 + l15;
      if (col >= N) continue;
      float bv = bias ? bias[col] : 0.0f;
#pragma unroll
      for (int i = 0; i < 8; ++i) {
        int row = m0 + wm + mt * 16 + rbase + i;
        if (row >= M) continue;
        float v = (acc[mt][nt][i] + bv) * scale;
        if (RELU) v = v > 0.f ? v : 0.f;
        if (OUT_F16) ((_Float16*)Cv)[(size_t)row * N + col] = (_Float16)v;
        else         ((float*)Cv)[(size_t)row * N + col] = v;
      }
    }
}

// ---------------------------------------------------------------------------
// Flash self-attention: 1 wave per 16-query tile per (b,h); D_HEAD=32.
// q,k,v f16 [(qi*B+b), h*32+d]; q pre-scaled by 1/sqrt(32). Output f16.
// ---------------------------------------------------------------------------
__global__ __launch_bounds__(32)
void flash_attn(const _Float16* __restrict__ q, const _Float16* __restrict__ k,
                const _Float16* __restrict__ v, _Float16* __restrict__ o)
{
  __shared__ _Float16 p_s[16][32];

  const int lane = threadIdx.x & 31;
  const int l15  = lane & 15;
  const int kb   = (lane >> 4) * 8;
  const int bh = blockIdx.y;
  const int b = bh >> 3, h = bh & 7;
  const int q0 = blockIdx.x * 16;

  auto IDX = [&](int tok, int d) { return ((size_t)tok * BB + b) * D_MODEL + h * D_HEAD + d; };

  HFrag qf;
  {
    int qi = q0 + l15; if (qi > LQ - 1) qi = LQ - 1;
    qf.h2[0] = *(const v8h*)&q[IDX(qi, kb)];
    qf.h2[1] = *(const v8h*)&q[IDX(qi, 16 + kb)];
  }

  v8f accd[2] = {};
  float rm[8], rl[8];
#pragma unroll
  for (int i = 0; i < 8; ++i) { rm[i] = -1e30f; rl[i] = 0.f; }

  for (int kt = 0; kt < LQ; kt += 32) {
    HFrag kf[2];
#pragma unroll
    for (int nt = 0; nt < 2; ++nt) {
      int key = kt + nt * 16 + l15; if (key > LQ - 1) key = LQ - 1;
      kf[nt].h2[0] = *(const v8h*)&k[IDX(key, kb)];
      kf[nt].h2[1] = *(const v8h*)&k[IDX(key, 16 + kb)];
    }
    v8f s0 = {}, s1 = {};
    s0 = wmma_f16(qf, kf[0], s0);
    s1 = wmma_f16(qf, kf[1], s1);

    const bool ok0 = (kt + l15) < LQ;
    const bool ok1 = (kt + 16 + l15) < LQ;
    float mx[8];
#pragma unroll
    for (int i = 0; i < 8; ++i) {
      float a = ok0 ? s0[i] : -1e30f;
      float c = ok1 ? s1[i] : -1e30f;
      s0[i] = a; s1[i] = c;
      mx[i] = fmaxf(a, c);
    }
#pragma unroll
    for (int i = 0; i < 8; ++i)
#pragma unroll
      for (int off = 1; off < 16; off <<= 1)
        mx[i] = fmaxf(mx[i], __shfl_xor(mx[i], off, 32));

    float corr[8], rs[8];
#pragma unroll
    for (int i = 0; i < 8; ++i) {
      float nm = fmaxf(rm[i], mx[i]);
      corr[i] = __expf(rm[i] - nm);
      rm[i] = nm;
      float p0 = __expf(s0[i] - nm);
      float p1 = __expf(s1[i] - nm);
      s0[i] = p0; s1[i] = p1;
      rs[i] = p0 + p1;
    }
#pragma unroll
    for (int i = 0; i < 8; ++i)
#pragma unroll
      for (int off = 1; off < 16; off <<= 1)
        rs[i] += __shfl_xor(rs[i], off, 32);
#pragma unroll
    for (int i = 0; i < 8; ++i) rl[i] = rl[i] * corr[i] + rs[i];
#pragma unroll
    for (int t = 0; t < 2; ++t)
#pragma unroll
      for (int i = 0; i < 8; ++i) accd[t][i] *= corr[i];

    const int rrow = (lane >> 4) * 8;
    __syncthreads();
#pragma unroll
    for (int i = 0; i < 8; ++i) {
      p_s[rrow + i][l15]      = (_Float16)s0[i];
      p_s[rrow + i][16 + l15] = (_Float16)s1[i];
    }
    __syncthreads();
    HFrag pf;
    pf.h2[0] = *(const v8h*)&p_s[l15][kb];
    pf.h2[1] = *(const v8h*)&p_s[l15][16 + kb];

    HFrag vf[2];
#pragma unroll
    for (int dt = 0; dt < 2; ++dt) {
      int d = dt * 16 + l15;
#pragma unroll
      for (int j = 0; j < 8; ++j) {
        int k0i = kt + kb + j;      if (k0i > LQ - 1) k0i = LQ - 1;
        int k1i = kt + 16 + kb + j; if (k1i > LQ - 1) k1i = LQ - 1;
        vf[dt].e[j]     = v[IDX(k0i, d)];
        vf[dt].e[8 + j] = v[IDX(k1i, d)];
      }
    }
    accd[0] = wmma_f16(pf, vf[0], accd[0]);
    accd[1] = wmma_f16(pf, vf[1], accd[1]);
  }

  const int rrow = (lane >> 4) * 8;
#pragma unroll
  for (int dt = 0; dt < 2; ++dt)
#pragma unroll
    for (int i = 0; i < 8; ++i) {
      int qo = q0 + rrow + i;
      if (qo >= LQ) continue;
      o[IDX(qo, dt * 16 + l15)] = (_Float16)(accd[dt][i] / rl[i]);
    }
}

// ---------------------------------------------------------------------------
// Small elementwise / conversion kernels
// ---------------------------------------------------------------------------
__global__ __launch_bounds__(256)
void conv_f16(const float* __restrict__ a, _Float16* __restrict__ out, int n)
{
  int i = blockIdx.x * 256 + threadIdx.x;
  if (i < n) out[i] = (_Float16)a[i];
}

__global__ __launch_bounds__(256)
void add_f16(const float* __restrict__ a, const float* __restrict__ b,
             _Float16* __restrict__ out, int n)
{
  int i = blockIdx.x * 256 + threadIdx.x;
  if (i < n) out[i] = (_Float16)(a[i] + b[i]);
}

// W[K][N] f32 -> Wt[N][K] f16 (LDS tile transpose, 32x32)
__global__ __launch_bounds__(256)
void transpose_to_f16(const float* __restrict__ W, _Float16* __restrict__ Wt,
                      int K, int N)
{
  __shared__ float t[32][33];
  const int tx = threadIdx.x & 31;
  const int ty = threadIdx.x >> 5;
  const int k0 = blockIdx.y * 32, n0 = blockIdx.x * 32;
#pragma unroll
  for (int j = 0; j < 4; ++j) {
    int kk = k0 + ty + j * 8, nn = n0 + tx;
    t[ty + j * 8][tx] = (kk < K && nn < N) ? W[(size_t)kk * N + nn] : 0.f;
  }
  __syncthreads();
#pragma unroll
  for (int j = 0; j < 4; ++j) {
    int nn = n0 + ty + j * 8, kk = k0 + tx;
    if (nn < N && kk < K) Wt[(size_t)nn * K + kk] = (_Float16)t[tx][ty + j * 8];
  }
}

// ---------------------------------------------------------------------------
// Fused residual LayerNorm: out = LN(a + d) * g + b. Optional f16 mirror.
// ---------------------------------------------------------------------------
template<bool EMIT_F16>
__global__ __launch_bounds__(256)
void ln_resid(const float* __restrict__ a, const float* __restrict__ d,
              const float* __restrict__ g, const float* __restrict__ bb,
              float* __restrict__ out, _Float16* __restrict__ out_h)
{
  __shared__ float red[8];
  const int row = blockIdx.x;
  const int t = threadIdx.x;
  const size_t base = (size_t)row * D_MODEL;
  float x = a[base + t] + d[base + t];

  float s = x;
#pragma unroll
  for (int off = 1; off < 32; off <<= 1) s += __shfl_xor(s, off, 32);
  if ((t & 31) == 0) red[t >> 5] = s;
  __syncthreads();
  float tot = 0.f;
#pragma unroll
  for (int i = 0; i < 8; ++i) tot += red[i];
  const float mean = tot * (1.0f / D_MODEL);
  __syncthreads();

  float c = x - mean;
  float s2 = c * c;
#pragma unroll
  for (int off = 1; off < 32; off <<= 1) s2 += __shfl_xor(s2, off, 32);
  if ((t & 31) == 0) red[t >> 5] = s2;
  __syncthreads();
  float tv = 0.f;
#pragma unroll
  for (int i = 0; i < 8; ++i) tv += red[i];
  const float var = tv * (1.0f / D_MODEL);

  float r = c * rsqrtf(var + 1e-5f) * g[t] + bb[t];
  out[base + t] = r;
  if (EMIT_F16) out_h[base + t] = (_Float16)r;
}

// ---------------------------------------------------------------------------
// Multi-scale deformable sampling. One wave per (b,q,h); lane = channel d.
// ---------------------------------------------------------------------------
__global__ __launch_bounds__(256)
void ms_deform(const float* __restrict__ ref, const float* __restrict__ off,
               const float* __restrict__ awl, const _Float16* __restrict__ value,
               _Float16* __restrict__ samp)
{
  const int lane = threadIdx.x & 31;
  const int w = blockIdx.x * 8 + (threadIdx.x >> 5);
  const int h = w & 7;
  const int q = (w >> 3) % LQ;
  const int b = w / (8 * LQ);

  const int HW[N_LEVELS][2] = {{128,128},{64,64},{32,32},{16,16},{8,8}};
  const int LS[N_LEVELS] = {0, 16384, 20480, 21504, 21760};

  const size_t tok = (size_t)q * BB + b;

  float wl[N_LEVELS * N_POINTS];
  float mxw = -1e30f;
#pragma unroll
  for (int j = 0; j < N_LEVELS * N_POINTS; ++j) {
    wl[j] = awl[tok * (H_HEADS * N_LEVELS * N_POINTS) + h * (N_LEVELS * N_POINTS) + j];
    mxw = fmaxf(mxw, wl[j]);
  }
  float sw = 0.f;
#pragma unroll
  for (int j = 0; j < N_LEVELS * N_POINTS; ++j) { wl[j] = __expf(wl[j] - mxw); sw += wl[j]; }
  const float inv = 1.0f / sw;

  float acc = 0.f;
#pragma unroll
  for (int l = 0; l < N_LEVELS; ++l) {
    const int Hh = HW[l][0], Ww = HW[l][1];
    const float rx = ref[(tok * N_LEVELS + l) * 2 + 0];
    const float ry = ref[(tok * N_LEVELS + l) * 2 + 1];
#pragma unroll
    for (int p = 0; p < N_POINTS; ++p) {
      const size_t obase = tok * (H_HEADS * N_LEVELS * N_POINTS * 2)
                         + h * (N_LEVELS * N_POINTS * 2) + l * (N_POINTS * 2) + p * 2;
      const float ox = off[obase + 0];
      const float oy = off[obase + 1];
      const float x = (rx + ox / (float)Ww) * (float)Ww - 0.5f;
      const float y = (ry + oy / (float)Hh) * (float)Hh - 0.5f;
      const float x0 = floorf(x), y0 = floorf(y);
      const float fx = x - x0, fy = y - y0;
      const int ix = (int)x0, iy = (int)y0;
      float sv = 0.f;
#pragma unroll
      for (int dy = 0; dy < 2; ++dy)
#pragma unroll
        for (int dx = 0; dx < 2; ++dx) {
          const int xc = ix + dx, yc = iy + dy;
          if (xc < 0 || xc >= Ww || yc < 0 || yc >= Hh) continue;
          const float cw = (dx ? fx : 1.f - fx) * (dy ? fy : 1.f - fy);
          const size_t s = (size_t)LS[l] + (size_t)yc * Ww + xc;
          sv += cw * (float)value[(s * BB + b) * D_MODEL + h * D_HEAD + lane];
        }
      acc += wl[l * N_POINTS + p] * inv * sv;
    }
  }
  samp[tok * D_MODEL + h * D_HEAD + lane] = (_Float16)acc;
}

// ---------------------------------------------------------------------------
// Host-side orchestration
// ---------------------------------------------------------------------------
static inline size_t alignup(size_t x) { return (x + 255) & ~(size_t)255; }

extern "C" void kernel_launch(void* const* d_in, const int* in_sizes, int n_in,
                              void* d_out, int out_size, void* d_ws, size_t ws_size,
                              hipStream_t stream) {
  const float* tgt   = (const float*)d_in[0];
  const float* qpos  = (const float*)d_in[1];
  const float* refp  = (const float*)d_in[2];
  const float* mem   = (const float*)d_in[3];
  const float* W_in[10]  = { (const float*)d_in[6],  (const float*)d_in[7],
                             (const float*)d_in[8],  (const float*)d_in[9],
                             (const float*)d_in[10], (const float*)d_in[11],
                             (const float*)d_in[12], (const float*)d_in[13],
                             (const float*)d_in[14], (const float*)d_in[15] };
  const int  W_K[10] = { 256,256,256,256, 256,256,256,256, 256, 2048 };
  const int  W_N[10] = { 256,256,256,256, 320,160,256,256, 2048, 256 };
  const float* sa_b_q   = (const float*)d_in[16];
  const float* sa_b_k   = (const float*)d_in[17];
  const float* sa_b_v   = (const float*)d_in[18];
  const float* sa_b_o   = (const float*)d_in[19];
  const float* ca_b_off = (const float*)d_in[20];
  const float* ca_b_att = (const float*)d_in[21];
  const float* ca_b_val = (const float*)d_in[22];
  const float* ca_b_out = (const float*)d_in[23];
  const float* ffn_b1   = (const float*)d_in[24];
  const float* ffn_b2   = (const float*)d_in[25];
  const float* ln1_g = (const float*)d_in[26];
  const float* ln1_b = (const float*)d_in[27];
  const float* ln2_g = (const float*)d_in[28];
  const float* ln2_b = (const float*)d_in[29];
  const float* ln3_g = (const float*)d_in[30];
  const float* ln3_b = (const float*)d_in[31];

  char* ws = (char*)d_ws;
  size_t off = 0;
  auto alloc = [&](size_t bytes) { char* p = ws + off; off += alignup(bytes); return p; };

  const size_t TOKF = (size_t)NTOK * D_MODEL;            // 1.84M elements
  const size_t MEMF = (size_t)LV * BB * D_MODEL;         // 44.7M elements

  // f16 weight buffers (transposed [N][K])
  _Float16* Wt[10];
  for (int i = 0; i < 10; ++i)
    Wt[i] = (_Float16*)alloc((size_t)W_K[i] * W_N[i] * sizeof(_Float16));

  _Float16* tgt_h   = (_Float16*)alloc(TOKF * 2);
  _Float16* mem_h   = (_Float16*)alloc(MEMF * 2);
  _Float16* qk_h    = (_Float16*)alloc(TOKF * 2);   // qk, later qb
  _Float16* q_h     = (_Float16*)alloc(TOKF * 2);
  _Float16* k_h     = (_Float16*)alloc(TOKF * 2);
  _Float16* v_h     = (_Float16*)alloc(TOKF * 2);
  _Float16* ao_h    = (_Float16*)alloc(TOKF * 2);   // attention output (f16)
  float*    obuf    = (float*)alloc(TOKF * 4);      // proj outputs for residual
  float*    tgt_a   = (float*)alloc(TOKF * 4);
  float*    offb    = (float*)alloc((size_t)NTOK * 320 * 4);
  float*    awb     = (float*)alloc((size_t)NTOK * 160 * 4);
  _Float16* valb    = (_Float16*)alloc(MEMF * 2);
  _Float16* samp_h  = (_Float16*)alloc(TOKF * 2);
  float*    tgt_b   = (float*)alloc(TOKF * 4);
  _Float16* tgt_b_h = (_Float16*)alloc(TOKF * 2);
  _Float16* hb      = (_Float16*)alloc((size_t)NTOK * D_FF * 2);
  (void)ws_size;

  const float qscale = 0.17677669529663687f;   // 1/sqrt(32)
  auto ggrid = [](int M, int N) { return dim3((N + 63) / 64, (M + 63) / 64); };

  // ---- one-time (per launch) weight transpose+convert, input conversions ----
  for (int i = 0; i < 10; ++i)
    transpose_to_f16<<<dim3((W_N[i] + 31) / 32, (W_K[i] + 31) / 32), 256, 0, stream>>>(
        W_in[i], Wt[i], W_K[i], W_N[i]);
  conv_f16<<<(int)(TOKF / 256), 256, 0, stream>>>(tgt, tgt_h, (int)TOKF);
  conv_f16<<<(int)(MEMF / 256), 256, 0, stream>>>(mem, mem_h, (int)MEMF);
  add_f16<<<(int)(TOKF / 256), 256, 0, stream>>>(tgt, qpos, qk_h, (int)TOKF);

  // ---- self attention ----
  gemm_wmma<true, false><<<ggrid(NTOK, 256), 128, 0, stream>>>(
      qk_h, Wt[0], sa_b_q, q_h, NTOK, 256, 256, qscale);
  gemm_wmma<true, false><<<ggrid(NTOK, 256), 128, 0, stream>>>(
      qk_h, Wt[1], sa_b_k, k_h, NTOK, 256, 256, 1.0f);
  gemm_wmma<true, false><<<ggrid(NTOK, 256), 128, 0, stream>>>(
      tgt_h, Wt[2], sa_b_v, v_h, NTOK, 256, 256, 1.0f);
  flash_attn<<<dim3((LQ + 15) / 16, BB * H_HEADS), 32, 0, stream>>>(q_h, k_h, v_h, ao_h);
  gemm_wmma<false, false><<<ggrid(NTOK, 256), 128, 0, stream>>>(
      ao_h, Wt[3], sa_b_o, obuf, NTOK, 256, 256, 1.0f);
  ln_resid<false><<<NTOK, 256, 0, stream>>>(tgt, obuf, ln2_g, ln2_b, tgt_a, nullptr);

  // ---- deformable cross attention ----
  add_f16<<<(int)(TOKF / 256), 256, 0, stream>>>(tgt_a, qpos, qk_h, (int)TOKF);  // qb
  gemm_wmma<false, false><<<ggrid(NTOK, 320), 128, 0, stream>>>(
      qk_h, Wt[4], ca_b_off, offb, NTOK, 256, 320, 1.0f);
  gemm_wmma<false, false><<<ggrid(NTOK, 160), 128, 0, stream>>>(
      qk_h, Wt[5], ca_b_att, awb, NTOK, 256, 160, 1.0f);
  gemm_wmma<true, false><<<ggrid(LV * BB, 256), 128, 0, stream>>>(
      mem_h, Wt[6], ca_b_val, valb, LV * BB, 256, 256, 1.0f);
  ms_deform<<<NTOK, 256, 0, stream>>>(refp, offb, awb, valb, samp_h);
  gemm_wmma<false, false><<<ggrid(NTOK, 256), 128, 0, stream>>>(
      samp_h, Wt[7], ca_b_out, obuf, NTOK, 256, 256, 1.0f);
  ln_resid<true><<<NTOK, 256, 0, stream>>>(tgt_a, obuf, ln1_g, ln1_b, tgt_b, tgt_b_h);

  // ---- FFN ----
  gemm_wmma<true, true><<<ggrid(NTOK, D_FF), 128, 0, stream>>>(
      tgt_b_h, Wt[8], ffn_b1, hb, NTOK, 256, D_FF, 1.0f);
  gemm_wmma<false, false><<<ggrid(NTOK, 256), 128, 0, stream>>>(
      hb, Wt[9], ffn_b2, obuf, NTOK, D_FF, 256, 1.0f);
  ln_resid<false><<<NTOK, 256, 0, stream>>>(tgt_b, obuf, ln3_g, ln3_b, (float*)d_out, nullptr);

  (void)out_size; (void)in_sizes; (void)n_in;
}
